// Attention_8031588844286
// MI455X (gfx1250) — compile-verified, matrix operand order checked by reading, not run on hardware
//
#include <hip/hip_runtime.h>
#include <hip/hip_bf16.h>

typedef __bf16 bf16;
typedef __attribute__((ext_vector_type(16))) __bf16 v16bf;
typedef __attribute__((ext_vector_type(8)))  __bf16 v8bf;
typedef __attribute__((ext_vector_type(8)))  float  v8f;

#define SEQ 2048
#define DMODEL 4096

__device__ __forceinline__ v8f zero8() {
  v8f z = {0.f,0.f,0.f,0.f,0.f,0.f,0.f,0.f};
  return z;
}

// Build a 16-element bf16 A/B fragment from two 16-byte chunks.
// ISA 16-bit A/B layout: elems 0..7 hold K = g*8 + 0..7, elems 8..15 hold
// K = 16 + g*8 + 0..7, where g = lane/16 and lane%16 selects the M/N row.
__device__ __forceinline__ v16bf ld_frag16(const bf16* p0, const bf16* p1) {
  union { v16bf v; v8bf h[2]; } u;
  u.h[0] = *(const v8bf*)p0;
  u.h[1] = *(const v8bf*)p1;
  return u.v;
}

// Async bf16 16-byte copy global -> LDS (CDNA5 GLOBAL_LOAD_ASYNC_TO_LDS_B128,
// tracked by ASYNCcnt). ldsoff = low 32 bits of the flat LDS address (the
// aperture truncation rule makes this the wave-relative LDS byte address).
__device__ __forceinline__ void async_copy_b128(unsigned int ldsoff, const void* g) {
  asm volatile("global_load_async_to_lds_b128 %0, %1, off"
               :: "v"(ldsoff), "v"((unsigned long long)(uintptr_t)g)
               : "memory");
}
__device__ __forceinline__ void wait_async0() {
  asm volatile("s_wait_asynccnt 0x0" ::: "memory");
}

// ---------------------------------------------------------------------------
// C(MxN) = A(MxK) * B(KxN).  A: float or bf16, B: fp32 weights; both are
// converted to bf16 while staging into LDS.  BM=BN=128, BK=32; 256 threads =
// 8 waves (4M x 2N), each wave does 2x4 v_wmma_f32_16x16x32_bf16 per K-step.
// Register double-buffering: next K-slab is fetched into VGPRs during the
// WMMA phase and committed to LDS after the barrier (hides global latency).
// ---------------------------------------------------------------------------
template<typename TA, typename TC>
__global__ __launch_bounds__(256) void gemm_wmma(
    const TA* __restrict__ A, const float* __restrict__ B, TC* __restrict__ C,
    int M, int N, int K)
{
  (void)M;
  constexpr int BM = 128, BN = 128, BK = 32, BKP = 40; // pad keeps 16B rows
  __shared__ __align__(16) bf16 lds_a[BM][BKP];        // M x K
  __shared__ __align__(16) bf16 lds_b[BN][BKP];        // N x K (B transposed)

  const int t    = threadIdx.x;
  const int lane = t & 31, wid = t >> 5;
  const int wm   = wid & 3, wn = wid >> 2;
  const int half = lane >> 4, l16 = lane & 15;
  const int mbase = blockIdx.y * BM, nbase = blockIdx.x * BN;

  const int am  = t >> 5;       // A row base (+ i*8), A col = ak
  const int ak  = t & 31;
  const int bn  = t & 127;      // B col (stored as LDS row)
  const int bk0 = t >> 7;       // B row base (+ i*2)

  float pa[16], pb[16];
  auto fetch = [&](int k0) {
    #pragma unroll
    for (int i = 0; i < 16; ++i)
      pa[i] = (float)A[(size_t)(mbase + am + i * 8) * K + k0 + ak];
    #pragma unroll
    for (int i = 0; i < 16; ++i)
      pb[i] = B[(size_t)(k0 + bk0 + i * 2) * N + nbase + bn];
  };
  auto stage = [&]() {
    #pragma unroll
    for (int i = 0; i < 16; ++i) lds_a[am + i * 8][ak] = (bf16)pa[i];
    #pragma unroll
    for (int i = 0; i < 16; ++i) lds_b[bn][bk0 + i * 2] = (bf16)pb[i];
  };

  fetch(0);
  stage();
  __syncthreads();

  v8f acc[2][4];
  #pragma unroll
  for (int i = 0; i < 2; ++i)
    #pragma unroll
    for (int j = 0; j < 4; ++j) acc[i][j] = zero8();

  for (int k0 = 0; k0 < K; k0 += BK) {
    const bool more = (k0 + BK) < K;
    if (more) fetch(k0 + BK);          // overlaps with WMMA phase below

    v16bf af[2], bfr[4];
    #pragma unroll
    for (int mt = 0; mt < 2; ++mt) {
      int row = wm * 32 + mt * 16 + l16;
      af[mt] = ld_frag16(&lds_a[row][half * 8], &lds_a[row][16 + half * 8]);
    }
    #pragma unroll
    for (int nt = 0; nt < 4; ++nt) {
      int col = wn * 64 + nt * 16 + l16;
      bfr[nt] = ld_frag16(&lds_b[col][half * 8], &lds_b[col][16 + half * 8]);
    }
    #pragma unroll
    for (int mt = 0; mt < 2; ++mt)
      #pragma unroll
      for (int nt = 0; nt < 4; ++nt)
        acc[mt][nt] = __builtin_amdgcn_wmma_f32_16x16x32_bf16(
            false, af[mt], false, bfr[nt], (short)0, acc[mt][nt], false, false);

    __syncthreads();                   // all waves done reading this slab
    if (more) stage();                 // commit prefetched slab
    __syncthreads();
  }

  // C layout: lane holds (row = r + 8*(lane/16), col = lane%16) in VGPR r.
  #pragma unroll
  for (int mt = 0; mt < 2; ++mt)
    #pragma unroll
    for (int nt = 0; nt < 4; ++nt)
      #pragma unroll
      for (int r = 0; r < 8; ++r) {
        int row = mbase + wm * 32 + mt * 16 + r + 8 * half;
        int col = nbase + wn * 64 + nt * 16 + l16;
        C[(size_t)row * N + col] = (TC)acc[mt][nt][r];
      }
}

// ---------------------------------------------------------------------------
// RoPE (interleaved even/odd pairs) + permute (S, nh*128) -> [nh][S][128].
// ---------------------------------------------------------------------------
__global__ __launch_bounds__(64) void rope_permute(
    const bf16* __restrict__ src, bf16* __restrict__ dst,
    const float* __restrict__ cosb, const float* __restrict__ sinb, int nh)
{
  const int s = blockIdx.x;
  const int h = blockIdx.y;
  const int i = threadIdx.x;            // pair index 0..63
  const size_t si = (size_t)s * (nh * 128) + h * 128 + 2 * i;
  float te = (float)src[si];
  float to = (float)src[si + 1];
  float c  = cosb[s * 64 + i];
  float sn = sinb[s * 64 + i];
  float oe = te * c - to * sn;
  float oo = te * sn + to * c;
  const size_t di = ((size_t)h * SEQ + s) * 128 + 2 * i;
  dst[di]     = (bf16)oe;
  dst[di + 1] = (bf16)oo;
}

// ---------------------------------------------------------------------------
// V: (S, 8*128) -> vt[h][d][s] (per-head transposed), tiled through LDS so
// both the global loads and the global stores are contiguous 16B chunks.
// Grid: (SEQ/64, 8), 256 threads.
// ---------------------------------------------------------------------------
__global__ __launch_bounds__(256) void transpose_v(
    const bf16* __restrict__ v_raw, bf16* __restrict__ vt)
{
  __shared__ __align__(16) bf16 tile[64][136];
  const int t  = threadIdx.x;
  const int h  = blockIdx.y;
  const int s0 = blockIdx.x * 64;
  #pragma unroll
  for (int i = 0; i < 4; ++i) {
    int c = t + i * 256;                // 1024 chunks: 64 s-rows x 16
    int s = c >> 4, k8 = (c & 15) * 8;
    *(v8bf*)&tile[s][k8] =
        *(const v8bf*)&v_raw[(size_t)(s0 + s) * 1024 + h * 128 + k8];
  }
  __syncthreads();
  #pragma unroll
  for (int i = 0; i < 4; ++i) {
    int c = t + i * 256;                // 1024 chunks: 128 d-rows x 8
    int d = c >> 3, j8 = (c & 7) * 8;
    v8bf o;
    #pragma unroll
    for (int e = 0; e < 8; ++e) o[e] = tile[j8 + e][d];
    *(v8bf*)&vt[((size_t)h * 128 + d) * SEQ + s0 + j8] = o;
  }
}

// ---------------------------------------------------------------------------
// Flash attention, one WG per (head, 128 query rows), 8 waves x 16 rows.
// K (j x d) and V^T (d x j) tiles staged with async global->LDS B128 copies;
// scores and PV via v_wmma_f32_16x16x32_bf16; online softmax with lane-local
// row stats (rows r + 8*(lane/16) live in the same lane across all accums).
// ---------------------------------------------------------------------------
__global__ __launch_bounds__(256) void attn_fa(
    const bf16* __restrict__ qr, const bf16* __restrict__ kr,
    const bf16* __restrict__ vt, bf16* __restrict__ attn)
{
  constexpr float SM_SCALE = 0.08838834764831845f;   // 1/sqrt(128)
  const int h  = blockIdx.x;
  const int qb = blockIdx.y * 128;
  const int hk = h >> 2;                 // N_REP = 4 (GQA)
  const int t  = threadIdx.x;
  const int lane = t & 31, w = t >> 5;
  const int half = lane >> 4, l16 = lane & 15;

  __shared__ __align__(16) bf16 lds_k [64][136];     // j x d  (N x K for QK^T)
  __shared__ __align__(16) bf16 lds_vt[128][72];     // d x j  (N x K for PV)
  __shared__ __align__(16) bf16 lds_p [8][16][72];   // per-wave P (16 x 64)

  const unsigned int kbase = (unsigned int)(uintptr_t)&lds_k[0][0];
  const unsigned int vbase = (unsigned int)(uintptr_t)&lds_vt[0][0];

  // Q fragments: wave w owns rows [qb + w*16, qb + w*16 + 16); loaded once.
  const bf16* qrow = qr + ((size_t)h * SEQ + qb + w * 16 + l16) * 128;
  v16bf qf[4];
  #pragma unroll
  for (int kf = 0; kf < 4; ++kf)
    qf[kf] = ld_frag16(qrow + kf * 32 + half * 8,
                       qrow + kf * 32 + 16 + half * 8);

  v8f oacc[8];
  #pragma unroll
  for (int i = 0; i < 8; ++i) oacc[i] = zero8();
  float mi[8], li[8];
  #pragma unroll
  for (int r = 0; r < 8; ++r) { mi[r] = -1e30f; li[r] = 0.f; }

  const int jb_end = (qb + 127) >> 6;     // causal: only blocks with j <= q_max
  for (int jb = 0; jb <= jb_end; ++jb) {
    // Async-stage K rows and V^T rows (pure bf16 b128 copies).
    #pragma unroll
    for (int i = 0; i < 4; ++i) {
      int c = t + i * 256;                // 1024 chunks: 64 j-rows x 16
      int j = c >> 4, k8 = (c & 15) * 8;
      async_copy_b128(kbase + (unsigned int)(j * 272 + k8 * 2),
                      kr + ((size_t)hk * SEQ + jb * 64 + j) * 128 + k8);
    }
    #pragma unroll
    for (int i = 0; i < 4; ++i) {
      int c = t + i * 256;                // 1024 chunks: 128 d-rows x 8
      int d = c >> 3, j8 = (c & 7) * 8;
      async_copy_b128(vbase + (unsigned int)(d * 144 + j8 * 2),
                      vt + ((size_t)hk * 128 + d) * SEQ + jb * 64 + j8);
    }
    wait_async0();
    __syncthreads();

    // Scores: S(16x64) = Q(16x128) . K^T, accumulated over 4 K-steps.
    v8f sacc[4];
    #pragma unroll
    for (int i = 0; i < 4; ++i) sacc[i] = zero8();
    #pragma unroll
    for (int kf = 0; kf < 4; ++kf)
      #pragma unroll
      for (int nt = 0; nt < 4; ++nt) {
        v16bf kfrag = ld_frag16(&lds_k[nt * 16 + l16][kf * 32 + half * 8],
                                &lds_k[nt * 16 + l16][kf * 32 + 16 + half * 8]);
        sacc[nt] = __builtin_amdgcn_wmma_f32_16x16x32_bf16(
            false, qf[kf], false, kfrag, (short)0, sacc[nt], false, false);
      }

    // Scale + causal mask + per-row running max (lane-local rows r + 8*half).
    float rmax[8];
    #pragma unroll
    for (int r = 0; r < 8; ++r) rmax[r] = -1e30f;
    #pragma unroll
    for (int nt = 0; nt < 4; ++nt) {
      int j = jb * 64 + nt * 16 + l16;
      #pragma unroll
      for (int r = 0; r < 8; ++r) {
        float v = sacc[nt][r] * SM_SCALE;
        int q = qb + w * 16 + r + 8 * half;
        if (j > q) v = -1e30f;
        sacc[nt][r] = v;
        rmax[r] = fmaxf(rmax[r], v);
      }
    }
    #pragma unroll
    for (int m = 1; m < 16; m <<= 1)
      #pragma unroll
      for (int r = 0; r < 8; ++r)
        rmax[r] = fmaxf(rmax[r], __shfl_xor(rmax[r], m, 32));

    // Online softmax update; rescale running output.
    float rsum[8];
    #pragma unroll
    for (int r = 0; r < 8; ++r) {
      float mn = fmaxf(mi[r], rmax[r]);
      float alpha = __expf(mi[r] - mn);
      mi[r] = mn; li[r] *= alpha; rsum[r] = 0.f;
      #pragma unroll
      for (int a = 0; a < 8; ++a) oacc[a][r] *= alpha;
    }
    #pragma unroll
    for (int nt = 0; nt < 4; ++nt)
      #pragma unroll
      for (int r = 0; r < 8; ++r) {
        float p = __expf(sacc[nt][r] - mi[r]);
        rsum[r] += p;
        lds_p[w][r + 8 * half][nt * 16 + l16] = (bf16)p;  // C- -> A-layout swap
      }
    #pragma unroll
    for (int m = 1; m < 16; m <<= 1)
      #pragma unroll
      for (int r = 0; r < 8; ++r) rsum[r] += __shfl_xor(rsum[r], m, 32);
    #pragma unroll
    for (int r = 0; r < 8; ++r) li[r] += rsum[r];

    // O += P(16x64) . V(64x128)  (per-wave LDS; in-order DS pipe, no barrier).
    #pragma unroll
    for (int js = 0; js < 2; ++js) {
      v16bf pf = ld_frag16(&lds_p[w][l16][js * 32 + half * 8],
                           &lds_p[w][l16][js * 32 + 16 + half * 8]);
      #pragma unroll
      for (int dt = 0; dt < 8; ++dt) {
        v16bf vf = ld_frag16(&lds_vt[dt * 16 + l16][js * 32 + half * 8],
                             &lds_vt[dt * 16 + l16][js * 32 + 16 + half * 8]);
        oacc[dt] = __builtin_amdgcn_wmma_f32_16x16x32_bf16(
            false, pf, false, vf, (short)0, oacc[dt], false, false);
      }
    }
    __syncthreads();                     // protect lds_k/lds_vt for next jb
  }

  float inv[8];
  #pragma unroll
  for (int r = 0; r < 8; ++r) inv[r] = 1.0f / li[r];
  #pragma unroll
  for (int dt = 0; dt < 8; ++dt)
    #pragma unroll
    for (int r = 0; r < 8; ++r) {
      int q = qb + w * 16 + r + 8 * half;
      int d = dt * 16 + l16;
      attn[(size_t)q * DMODEL + h * 128 + d] = (bf16)(oacc[dt][r] * inv[r]);
    }
}

// ---------------------------------------------------------------------------
extern "C" void kernel_launch(void* const* d_in, const int* in_sizes, int n_in,
                              void* d_out, int out_size, void* d_ws, size_t ws_size,
                              hipStream_t stream) {
  (void)in_sizes; (void)n_in; (void)out_size; (void)ws_size;
  const float* x    = (const float*)d_in[0];
  const float* wq   = (const float*)d_in[1];
  const float* wk   = (const float*)d_in[2];
  const float* wv   = (const float*)d_in[3];
  const float* wo   = (const float*)d_in[4];
  const float* cosb = (const float*)d_in[5];
  const float* sinb = (const float*)d_in[6];
  // d_in[7] = mask: causal mask applied analytically in attn_fa.
  float* out = (float*)d_out;

  // Workspace layout (bf16 elements), ~50 MB total.
  bf16* ws    = (bf16*)d_ws;
  bf16* q_raw = ws;                                  // 2048*4096
  bf16* k_raw = q_raw + (size_t)SEQ * 4096;          // 2048*1024
  bf16* v_raw = k_raw + (size_t)SEQ * 1024;          // 2048*1024
  bf16* q_rot = v_raw + (size_t)SEQ * 1024;          // 32*2048*128
  bf16* k_rot = q_rot + (size_t)SEQ * 4096;          // 8*2048*128
  bf16* vt    = k_rot + (size_t)SEQ * 1024;          // 8*128*2048 (transposed)
  bf16* attn  = q_raw;  // q_raw is dead after RoPE; reuse (stream-ordered)

  dim3 blk(256);
  gemm_wmma<float, bf16><<<dim3(32, 16), blk, 0, stream>>>(x, wq, q_raw, SEQ, 4096, 4096);
  gemm_wmma<float, bf16><<<dim3( 8, 16), blk, 0, stream>>>(x, wk, k_raw, SEQ, 1024, 4096);
  gemm_wmma<float, bf16><<<dim3( 8, 16), blk, 0, stream>>>(x, wv, v_raw, SEQ, 1024, 4096);
  rope_permute<<<dim3(SEQ, 32), 64, 0, stream>>>(q_raw, q_rot, cosb, sinb, 32);
  rope_permute<<<dim3(SEQ,  8), 64, 0, stream>>>(k_raw, k_rot, cosb, sinb,  8);
  transpose_v<<<dim3(SEQ / 64, 8), blk, 0, stream>>>(v_raw, vt);
  attn_fa<<<dim3(32, 16), blk, 0, stream>>>(q_rot, k_rot, vt, attn);
  gemm_wmma<bf16, float><<<dim3(32, 16), blk, 0, stream>>>(attn, wo, out, SEQ, 4096, 4096);
}
